// GCNdiff_36515811951274
// MI455X (gfx1250) — compile-verified, matrix operand order checked by reading, not run on hardware
//
#include <hip/hip_runtime.h>
#include <math.h>

// ---------------------------------------------------------------------------
// CDNA5 / gfx1250 implementation of the GCN-diffusion forward pass.
// All matmul-like work (linears, cheb-conv passes, im2col'd temporal convs,
// attention projections) funnels through one WMMA GEMM kernel using
// v_wmma_f32_16x16x32_f16 (f32 accumulate, f16 tiles staged in LDS in
// fragment order, vectorized interior fast path + prefetch of next K tile).
// Irregular ops (top-k, softmax attention, layernorms, gathers) are VALU.
// ---------------------------------------------------------------------------

typedef __attribute__((ext_vector_type(16))) _Float16 v16h;
typedef __attribute__((ext_vector_type(4)))  _Float16 v4h;
typedef __attribute__((ext_vector_type(8)))  float    v8f;

#define N_PTS   17
#define HIDD    96
#define EMDD    384
#define HEADS   4
#define HDIM    24            // 96/4
#define LAYERS  5
#define PASTF   6
#define FEATC   64
#define K_NEAR  50
#define THRE    0.04f
#define BATCH   256
#define ROWS    (BATCH*N_PTS)        // 4352
#define LROWS   (ROWS*K_NEAR)        // 217600
#define THIDC   (HIDD*N_PTS)         // 1632
#define RPROWS  (BATCH*PASTF)        // 1536
#define KIM     (3*THIDC)            // 4896
#define RADAR_N 4096
#define XDIM    21                   // 3*(PAST+1)
#define LIMB    16
#define LIMBF   1024
#define LLD     (N_PTS*FEATC)        // 1088

// Position of K-index c (0..31) inside a lane's 16-half WMMA fragment,
// per the documented 16-bit A layout (VGPR j<4: K=kb+2j; j>=4: K=16+kb+2(j-4)).
__device__ __forceinline__ int fragpos(int c) {
  int j = (c & 16) ? (4 + ((c & 7) >> 1)) : ((c & 7) >> 1);
  return 2 * j + (c & 1);
}

// ---------------------------------------------------------------------------
// WMMA GEMM: C = act(A(MxK) @ B(KxN) + bias [+ C if accum])
// 256 threads = 8 waves (2 M-groups x 4 N-groups); block tile 64x64;
// each wave: 32x16 output = 2 WMMAs per 32-K step sharing one B fragment.
// act: 0=none, 1=relu, 2=silu
// ---------------------------------------------------------------------------
__global__ void __launch_bounds__(256)
gemm_wmma_kernel(const float* __restrict__ A, const float* __restrict__ B,
                 const float* __restrict__ bias, float* __restrict__ C,
                 int M, int N, int K, int lda, int ldb, int ldc,
                 int accum, int act)
{
  // Tiles staged directly in fragment order: [subtile][lane][16 halves]
  __shared__ __align__(32) _Float16 Af[4][32][16];   // 4 x 16-row M subtiles
  __shared__ __align__(32) _Float16 Bf[4][32][16];   // 4 x 16-col N subtiles
  const int tid  = threadIdx.x;
  const int wave = tid >> 5, lane = tid & 31;
  const int mg = wave >> 2;          // 0..1 : 32-row slab
  const int ng = wave & 3;           // 0..3 : 16-col slab
  const int m0 = blockIdx.y << 6;
  const int n0 = blockIdx.x << 6;

  const v8f vz = {0.f,0.f,0.f,0.f,0.f,0.f,0.f,0.f};
  v8f accv[2]; accv[0] = vz; accv[1] = vz;

  // fast-path per-thread chunk coordinates (2 x float4 per thread per tile)
  const int arow0 = (tid * 4) >> 5,          acol0 = (tid * 4) & 31;
  const int arow1 = ((tid + 256) * 4) >> 5,  acol1 = ((tid + 256) * 4) & 31;
  const int brow0 = (tid * 4) >> 6,          bcol0 = (tid * 4) & 63;
  const int brow1 = ((tid + 256) * 4) >> 6,  bcol1 = ((tid + 256) * 4) & 63;
  const bool aAligned = ((lda & 3) == 0);
  const bool bAligned = ((ldb & 3) == 0);

  for (int k0 = 0; k0 < K; k0 += 32) {
    const bool aFast = aAligned && (k0 + 32 <= K) && (m0 + 64 <= M);
    const bool bFast = bAligned && (k0 + 32 <= K) && (n0 + 64 <= N);

    if (aFast) {
      {
        const float4 f = *(const float4*)(A + (size_t)(m0 + arow0) * lda + k0 + acol0);
        v4h h; h.x = (_Float16)f.x; h.y = (_Float16)f.y; h.z = (_Float16)f.z; h.w = (_Float16)f.w;
        *(v4h*)&Af[arow0 >> 4][(arow0 & 15) + ((acol0 & 8) ? 16 : 0)][fragpos(acol0)] = h;
      }
      {
        const float4 f = *(const float4*)(A + (size_t)(m0 + arow1) * lda + k0 + acol1);
        v4h h; h.x = (_Float16)f.x; h.y = (_Float16)f.y; h.z = (_Float16)f.z; h.w = (_Float16)f.w;
        *(v4h*)&Af[arow1 >> 4][(arow1 & 15) + ((acol1 & 8) ? 16 : 0)][fragpos(acol1)] = h;
      }
      if (k0 + 64 <= K) {  // prefetch next K tile (global_prefetch_b8)
        __builtin_prefetch(A + (size_t)(m0 + arow0) * lda + k0 + 32 + acol0, 0, 1);
        __builtin_prefetch(A + (size_t)(m0 + arow1) * lda + k0 + 32 + acol1, 0, 1);
      }
    } else {
      for (int i = tid; i < 2048; i += 256) {
        int r = i >> 5, c = i & 31, gm = m0 + r, gk = k0 + c;
        float v = (gm < M && gk < K) ? A[(size_t)gm * lda + gk] : 0.f;
        Af[r >> 4][(r & 15) + ((c & 8) ? 16 : 0)][fragpos(c)] = (_Float16)v;
      }
    }

    if (bFast) {
      {
        const float4 f = *(const float4*)(B + (size_t)(k0 + brow0) * ldb + n0 + bcol0);
        const int lb = ((brow0 & 8) ? 16 : 0), p = fragpos(brow0);
        const int t = bcol0 >> 4, nl = bcol0 & 15;
        Bf[t][nl + 0 + lb][p] = (_Float16)f.x;
        Bf[t][nl + 1 + lb][p] = (_Float16)f.y;
        Bf[t][nl + 2 + lb][p] = (_Float16)f.z;
        Bf[t][nl + 3 + lb][p] = (_Float16)f.w;
      }
      {
        const float4 f = *(const float4*)(B + (size_t)(k0 + brow1) * ldb + n0 + bcol1);
        const int lb = ((brow1 & 8) ? 16 : 0), p = fragpos(brow1);
        const int t = bcol1 >> 4, nl = bcol1 & 15;
        Bf[t][nl + 0 + lb][p] = (_Float16)f.x;
        Bf[t][nl + 1 + lb][p] = (_Float16)f.y;
        Bf[t][nl + 2 + lb][p] = (_Float16)f.z;
        Bf[t][nl + 3 + lb][p] = (_Float16)f.w;
      }
      if (k0 + 64 <= K) {
        __builtin_prefetch(B + (size_t)(k0 + 32 + brow0) * ldb + n0 + bcol0, 0, 1);
        __builtin_prefetch(B + (size_t)(k0 + 32 + brow1) * ldb + n0 + bcol1, 0, 1);
      }
    } else {
      for (int i = tid; i < 2048; i += 256) {
        int k = i >> 6, n = i & 63, gk = k0 + k, gn = n0 + n;
        float v = (gk < K && gn < N) ? B[(size_t)gk * ldb + gn] : 0.f;
        Bf[n >> 4][(n & 15) + ((k & 8) ? 16 : 0)][fragpos(k)] = (_Float16)v;
      }
    }
    __syncthreads();

    const v16h av0 = *(const v16h*)Af[mg * 2 + 0][lane];
    const v16h av1 = *(const v16h*)Af[mg * 2 + 1][lane];
    const v16h bv  = *(const v16h*)Bf[ng][lane];
#pragma unroll
    for (int sub = 0; sub < 2; ++sub)
      accv[sub] = __builtin_amdgcn_wmma_f32_16x16x32_f16(
          false, sub ? av1 : av0, false, bv, (short)0, accv[sub], false, false);
    __syncthreads();
  }

  const int gn = n0 + (ng << 4) + (lane & 15);
  if (gn < N) {
#pragma unroll
    for (int sub = 0; sub < 2; ++sub) {
#pragma unroll
      for (int r = 0; r < 8; ++r) {
        int gm = m0 + (mg << 5) + (sub << 4) + r + ((lane >> 4) << 3);
        if (gm < M) {
          float v = accv[sub][r];
          if (bias)  v += bias[gn];
          if (accum) v += C[(size_t)gm * ldc + gn];
          if (act == 1)      v = fmaxf(v, 0.f);
          else if (act == 2) v = v / (1.f + __expf(-v));
          C[(size_t)gm * ldc + gn] = v;
        }
      }
    }
  }
}

// --------------------------- small VALU kernels ----------------------------

__global__ void build_L_kernel(const float* __restrict__ adj, float* __restrict__ L) {
  __shared__ float dinv[N_PTS];
  int tid = threadIdx.x;
  if (tid < N_PTS) {
    float s = 0.f;
    for (int m = 0; m < N_PTS; ++m) s += adj[tid * N_PTS + m];
    dinv[tid] = rsqrtf(s);
  }
  __syncthreads();
  if (tid < N_PTS * N_PTS) {
    int n = tid / N_PTS, m = tid % N_PTS;
    L[tid] = ((n == m) ? 1.f : 0.f) - dinv[n] * adj[tid] * dinv[m];
  }
}

__global__ void tsembed_kernel(const int* __restrict__ t, float* __restrict__ out) {
  int b = blockIdx.x, j = threadIdx.x;
  const int half = HIDD / 2;
  float tv = (float)t[b];
  int jj = (j < half) ? j : (j - half);
  float f = __expf(-logf(10000.f) * (float)jj / (float)(half - 1));
  float e = tv * f;
  out[b * HIDD + j] = (j < half) ? sinf(e) : cosf(e);
}

// out[g,n,c] = sum_m L[n,m] * in[g,m,c]   (graph Laplacian apply)
__global__ void lx_kernel(const float* __restrict__ in, float* __restrict__ out,
                          const float* __restrict__ L, int G, int C) {
  int idx = blockIdx.x * blockDim.x + threadIdx.x;
  int total = G * N_PTS * C;
  if (idx >= total) return;
  int c = idx % C, r = (idx / C) % N_PTS, g = idx / (C * N_PTS);
  float s = 0.f;
  for (int m = 0; m < N_PTS; ++m)
    s += L[r * N_PTS + m] * in[((size_t)g * N_PTS + m) * C + c];
  out[idx] = s;
}

// Per (batch,joint): select 50 nearest radar points + proximity scale.
__global__ void __launch_bounds__(128)
topk_kernel(const float* __restrict__ x, const float* __restrict__ radar,
            int* __restrict__ sel, float* __restrict__ scaleArr) {
  int bn = blockIdx.x;
  int b = bn / N_PTS, n = bn % N_PTS;
  __shared__ float dist[RADAR_N];
  __shared__ float rmin[128];
  __shared__ int   ridx[128];
  __shared__ int   s_cnt;
  float ax = x[((size_t)b * N_PTS + n) * XDIM + 18];
  float ay = x[((size_t)b * N_PTS + n) * XDIM + 19];
  float az = x[((size_t)b * N_PTS + n) * XDIM + 20];
  const float* pc = radar + (size_t)b * RADAR_N * 6;
  for (int p = threadIdx.x; p < RADAR_N; p += 128) {
    float dx = pc[p * 6 + 0] - ax, dy = pc[p * 6 + 1] - ay, dz = pc[p * 6 + 2] - az;
    dist[p] = dx * dx + dy * dy + dz * dz;
  }
  if (threadIdx.x == 0) s_cnt = 0;
  __syncthreads();
  for (int it = 0; it < K_NEAR; ++it) {
    float best = 3.4e38f; int bi = 0;
    for (int p = threadIdx.x; p < RADAR_N; p += 128)
      if (dist[p] < best) { best = dist[p]; bi = p; }
    rmin[threadIdx.x] = best; ridx[threadIdx.x] = bi;
    __syncthreads();
    for (int s2 = 64; s2 > 0; s2 >>= 1) {
      if (threadIdx.x < s2 && rmin[threadIdx.x + s2] < rmin[threadIdx.x]) {
        rmin[threadIdx.x] = rmin[threadIdx.x + s2];
        ridx[threadIdx.x] = ridx[threadIdx.x + s2];
      }
      __syncthreads();
    }
    if (threadIdx.x == 0) {
      int w = ridx[0];
      sel[(size_t)bn * K_NEAR + it] = w;
      if (rmin[0] < THRE) s_cnt++;
      dist[w] = 3.4e38f;
    }
    __syncthreads();
  }
  if (threadIdx.x == 0) scaleArr[bn] = (float)s_cnt / (float)K_NEAR;
}

__global__ void gather_pcs_kernel(const float* __restrict__ radar,
                                  const int* __restrict__ sel, float* __restrict__ pcs) {
  int idx = blockIdx.x * blockDim.x + threadIdx.x;
  if (idx >= LROWS * 6) return;
  int c = idx % 6, row = idx / 6, b = row / (N_PTS * K_NEAR);
  pcs[idx] = radar[((size_t)b * RADAR_N + sel[row]) * 6 + c];
}

// Softmax attention: one thread per query row, one block per (seq, head).
__global__ void __launch_bounds__(64)
attn_kernel(const float* __restrict__ q, const float* __restrict__ k,
            const float* __restrict__ v, float* __restrict__ o, int T) {
  int s = blockIdx.x, h = blockIdx.y, row = threadIdx.x;
  if (row >= T) return;
  const float* qp = q + ((size_t)s * T + row) * HIDD + h * HDIM;
  const float* kb = k + (size_t)s * T * HIDD + h * HDIM;
  const float* vb = v + (size_t)s * T * HIDD + h * HDIM;
  float qv[HDIM];
  for (int c = 0; c < HDIM; ++c) qv[c] = qp[c];
  float sc[K_NEAR];
  float scale = rsqrtf((float)HDIM);
  float mx = -3.4e38f;
  for (int j = 0; j < T; ++j) {
    const float* kp = kb + (size_t)j * HIDD;
    float a = 0.f;
    for (int c = 0; c < HDIM; ++c) a += qv[c] * kp[c];
    a *= scale; sc[j] = a; if (a > mx) mx = a;
  }
  float sum = 0.f;
  for (int j = 0; j < T; ++j) { sc[j] = __expf(sc[j] - mx); sum += sc[j]; }
  float inv = 1.f / sum;
  float ov[HDIM];
  for (int c = 0; c < HDIM; ++c) ov[c] = 0.f;
  for (int j = 0; j < T; ++j) {
    float p = sc[j] * inv;
    const float* vp = vb + (size_t)j * HIDD;
    for (int c = 0; c < HDIM; ++c) ov[c] += p * vp[c];
  }
  float* op = o + ((size_t)s * T + row) * HIDD + h * HDIM;
  for (int c = 0; c < HDIM; ++c) op[c] = ov[c];
}

__global__ void meanscale_kernel(const float* __restrict__ le,
                                 const float* __restrict__ scale, float* __restrict__ out) {
  int idx = blockIdx.x * blockDim.x + threadIdx.x;
  if (idx >= ROWS * HIDD) return;
  int h = idx % HIDD, row = idx / HIDD;
  float s = 0.f;
  for (int t = 0; t < K_NEAR; ++t) s += le[((size_t)row * K_NEAR + t) * HIDD + h];
  out[idx] = s * (1.f / (float)K_NEAR) * scale[row];
}

__global__ void frames_kernel(const float* __restrict__ x, float* __restrict__ fr) {
  int idx = blockIdx.x * blockDim.x + threadIdx.x;
  if (idx >= BATCH * PASTF * N_PTS * 3) return;
  int c = idx % 3, n = (idx / 3) % N_PTS, p = (idx / (3 * N_PTS)) % PASTF;
  int b = idx / (3 * N_PTS * PASTF);
  fr[idx] = x[((size_t)b * N_PTS + n) * XDIM + 3 * p + c];
}

// im2col over time (kernel 3, pad 1): in R x C -> out R x 3C
__global__ void im2col_kernel(const float* __restrict__ in, float* __restrict__ out,
                              int R, int C) {
  int idx = blockIdx.x * blockDim.x + threadIdx.x;
  if (idx >= R * 3 * C) return;
  int col = idx % (3 * C), row = idx / (3 * C);
  int dt = col / C, i = col % C, t = row % PASTF, tt = t + dt - 1;
  out[idx] = (tt >= 0 && tt < PASTF) ? in[(size_t)(row - t + tt) * C + i] : 0.f;
}

// (O,I,3) conv weight -> (3I, O) GEMM weight, K-index = dt*I + i
__global__ void repack_w_kernel(const float* __restrict__ w, float* __restrict__ wp) {
  int idx = blockIdx.x * blockDim.x + threadIdx.x;
  if (idx >= THIDC * THIDC * 3) return;
  int dt = idx % 3, i = (idx / 3) % THIDC, o = idx / (3 * THIDC);
  wp[((size_t)(dt * THIDC + i)) * THIDC + o] = w[idx];
}

__global__ void maxtime_kernel(const float* __restrict__ pe, float* __restrict__ out) {
  int idx = blockIdx.x * blockDim.x + threadIdx.x;
  if (idx >= ROWS * HIDD) return;
  int h = idx % HIDD, row = idx / HIDD, b = row / N_PTS, n = row % N_PTS;
  int ch = n * HIDD + h;
  float m = -3.4e38f;
  for (int t = 0; t < PASTF; ++t)
    m = fmaxf(m, pe[(size_t)(b * PASTF + t) * THIDC + ch]);
  out[idx] = m;
}

// graph layernorm: std with ddof=1, eps 1e-6 on std
__global__ void ln_gra_kernel(const float* __restrict__ x, const float* __restrict__ g,
                              const float* __restrict__ b, float* __restrict__ y, int D) {
  int row = blockIdx.x, lane = threadIdx.x;
  const float* xr = x + (size_t)row * D;
  float s = 0.f;
  for (int c = lane; c < D; c += 32) s += xr[c];
  for (int o = 16; o > 0; o >>= 1) s += __shfl_xor(s, o, 32);
  float mu = s / (float)D;
  float v = 0.f;
  for (int c = lane; c < D; c += 32) { float d = xr[c] - mu; v += d * d; }
  for (int o = 16; o > 0; o >>= 1) v += __shfl_xor(v, o, 32);
  float inv = 1.f / (sqrtf(v / (float)(D - 1)) + 1e-6f);
  float* yr = y + (size_t)row * D;
  for (int c = lane; c < D; c += 32) yr[c] = g[c] * (xr[c] - mu) * inv + b[c];
}

// torch layernorm: var ddof=0, eps 1e-5 inside rsqrt
__global__ void ln_torch_kernel(const float* __restrict__ x, const float* __restrict__ g,
                                const float* __restrict__ b, float* __restrict__ y, int D) {
  int row = blockIdx.x, lane = threadIdx.x;
  const float* xr = x + (size_t)row * D;
  float s = 0.f;
  for (int c = lane; c < D; c += 32) s += xr[c];
  for (int o = 16; o > 0; o >>= 1) s += __shfl_xor(s, o, 32);
  float mu = s / (float)D;
  float v = 0.f;
  for (int c = lane; c < D; c += 32) { float d = xr[c] - mu; v += d * d; }
  for (int o = 16; o > 0; o >>= 1) v += __shfl_xor(v, o, 32);
  float inv = rsqrtf(v / (float)D + 1e-5f);
  float* yr = y + (size_t)row * D;
  for (int c = lane; c < D; c += 32) yr[c] = g[c] * (xr[c] - mu) * inv + b[c];
}

__global__ void add3_kernel(float* __restrict__ out, const float* __restrict__ a,
                            const float* __restrict__ b, const float* __restrict__ c, int n) {
  int i = blockIdx.x * blockDim.x + threadIdx.x;
  if (i < n) out[i] = a[i] + b[i] + c[i];
}
__global__ void addin_kernel(float* __restrict__ out, const float* __restrict__ a, int n) {
  int i = blockIdx.x * blockDim.x + threadIdx.x;
  if (i < n) out[i] += a[i];
}
__global__ void copy_kernel(float* __restrict__ out, const float* __restrict__ a, int n) {
  int i = blockIdx.x * blockDim.x + threadIdx.x;
  if (i < n) out[i] = a[i];
}
__global__ void silu_sum_kernel(float* __restrict__ out, const float* __restrict__ a,
                                const float* __restrict__ b, int n) {
  int i = blockIdx.x * blockDim.x + threadIdx.x;
  if (i < n) { float v = a[i] + b[i]; out[i] = v / (1.f + __expf(-v)); }
}
// h += tvec[batch] (broadcast over joints) + cond
__global__ void addbc_kernel(float* __restrict__ h, const float* __restrict__ tvec,
                             const float* __restrict__ cond, int n) {
  int i = blockIdx.x * blockDim.x + threadIdx.x;
  if (i >= n) return;
  int c = i % HIDD, row = i / HIDD, b = row / N_PTS;
  h[i] += tvec[(size_t)b * HIDD + c] + cond[i];
}
__global__ void xlast_kernel(const float* __restrict__ x, float* __restrict__ xl) {
  int i = blockIdx.x * blockDim.x + threadIdx.x;
  if (i >= ROWS * 3) return;
  int c = i % 3, row = i / 3;
  xl[i] = x[(size_t)row * XDIM + 18 + c];
}

// ------------------------------- host side ---------------------------------

// param leaf indices (dict insertion order of params in setup_inputs)
enum {
  P_GIN1_W = 0, P_GIN1_B, P_GIN2_W, P_GIN2_B, P_GINT_W, P_GINT_B,
  P_TEMB_W0, P_TEMB_B0, P_TEMB_W1, P_TEMB_B1,
  P_LOC_W0, P_LOC_B0, P_LOC_W1, P_LOC_B1,
  P_LA_W, P_LA_B, P_TC1_W, P_TC1_B, P_TC2_W, P_TC2_B,
  P_LL_G, P_LL_B, P_LL_W0, P_LL_B0, P_LL_W1, P_LL_B1,
  P_LE_W0, P_LE_B0, P_LE_W1, P_LE_B1,
  P_AW, P_AB, P_LN1_G, P_LN1_B, P_LN2_G, P_LN2_B,
  P_GCN_W, P_GCN_B, P_G1_W, P_G1_B, P_G2_W, P_G2_B,
  P_TP_W, P_TP_B, P_GO_W, P_GO_B, P_COUNT
};

static inline void gemm(hipStream_t s, const float* A, const float* B, const float* bias,
                        float* C, int M, int N, int K, int lda, int ldb, int ldc,
                        int accum, int act) {
  dim3 grid((N + 63) / 64, (M + 63) / 64);
  gemm_wmma_kernel<<<grid, 256, 0, s>>>(A, B, bias, C, M, N, K, lda, ldb, ldc, accum, act);
}

#define EL(kern, total, ...) \
  kern<<<((total) + 255) / 256, 256, 0, stream>>>(__VA_ARGS__)

extern "C" void kernel_launch(void* const* d_in, const int* in_sizes, int n_in,
                              void* d_out, int out_size, void* d_ws, size_t ws_size,
                              hipStream_t stream) {
  (void)in_sizes; (void)n_in; (void)out_size; (void)ws_size;
  const float* x      = (const float*)d_in[0];
  // d_in[1] = mask (all ones -> no-op)
  const int*   tsteps = (const int*)d_in[2];
  const float* cemd   = (const float*)d_in[3];
  const float* radar  = (const float*)d_in[4];
  const float* limb   = (const float*)d_in[5];
  const float* adj    = (const float*)d_in[6 + P_COUNT];   // index 52
  auto P = [&](int i) { return (const float*)d_in[6 + i]; };

  // -------- workspace carving (floats, 256B-aligned slices) --------
  float* W = (float*)d_ws;
  size_t off = 0;
  auto alloc = [&](size_t n) {
    off = (off + 63) & ~size_t(63);
    float* p = W + off; off += n; return p;
  };
  float* Lm        = alloc(N_PTS * N_PTS);
  float* temb0     = alloc(BATCH * HIDD);
  float* tembH     = alloc(BATCH * EMDD);
  float* temb      = alloc(BATCH * EMDD);
  float* lembH     = alloc(BATCH * EMDD);
  float* lemb      = alloc(BATCH * EMDD);
  float* st        = alloc(BATCH * EMDD);        // silu(temb + lemb)
  float* cemdL     = alloc((size_t)BATCH * N_PTS * FEATC);
  float* globalE   = alloc((size_t)ROWS * HIDD);
  float* scaleArr  = alloc(ROWS);
  int*   selIdx    = (int*)alloc((size_t)ROWS * K_NEAR);
  float* pcs       = alloc((size_t)LROWS * 6);
  float* leh       = alloc((size_t)LROWS * 48);
  float* le        = alloc((size_t)LROWS * HIDD);
  float* qb        = alloc((size_t)LROWS * HIDD);
  float* kbuf      = alloc((size_t)LROWS * HIDD);
  float* vb        = alloc((size_t)LROWS * HIDD);
  float* ao        = alloc((size_t)LROWS * HIDD);
  float* localE    = alloc((size_t)ROWS * HIDD);
  float* frames    = alloc((size_t)RPROWS * N_PTS * 3);
  float* framesL   = alloc((size_t)RPROWS * N_PTS * 3);
  float* xp        = alloc((size_t)RPROWS * THIDC);     // == (RPROWS*N_PTS) x HID
  float* colb      = alloc((size_t)RPROWS * KIM);
  float* wp        = alloc((size_t)KIM * THIDC);        // reused for both convs
  float* pe1       = alloc((size_t)RPROWS * THIDC);
  float* pe2       = alloc((size_t)RPROWS * THIDC);
  float* pastE     = alloc((size_t)ROWS * HIDD);
  float* cond      = alloc((size_t)ROWS * HIDD);
  float* xlast     = alloc((size_t)ROWS * 3);
  float* xlastL    = alloc((size_t)ROWS * 3);
  float* outb      = alloc((size_t)ROWS * HIDD);
  float* hbuf      = alloc((size_t)ROWS * HIDD);
  float* h2        = alloc((size_t)ROWS * HIDD);
  float* Lbuf      = alloc((size_t)ROWS * HIDD);
  float* tvec      = alloc(BATCH * HIDD);
  float* lnC       = alloc((size_t)BATCH * LLD);
  float* llh       = alloc((size_t)BATCH * LIMBF);

  float* out_pose = (float*)d_out;                 // (256,17,21)
  float* out_limb = (float*)d_out + ROWS * XDIM;   // (256,16)

  // -------- 0) Laplacian (Tk[0]=I, Tk[1]=L) --------
  build_L_kernel<<<1, 320, 0, stream>>>(adj, Lm);

  // -------- 1) timestep + limb embeddings --------
  tsembed_kernel<<<BATCH, HIDD, 0, stream>>>(tsteps, temb0);
  gemm(stream, temb0, P(P_TEMB_W0), P(P_TEMB_B0), tembH, BATCH, EMDD, HIDD, HIDD, EMDD, EMDD, 0, 2);
  gemm(stream, tembH, P(P_TEMB_W1), P(P_TEMB_B1), temb,  BATCH, EMDD, EMDD, EMDD, EMDD, EMDD, 0, 0);
  gemm(stream, limb,  P(P_LE_W0),   P(P_LE_B0),   lembH, BATCH, EMDD, LIMB, LIMB, EMDD, EMDD, 0, 2);
  gemm(stream, lembH, P(P_LE_W1),   P(P_LE_B1),   lemb,  BATCH, EMDD, EMDD, EMDD, EMDD, EMDD, 0, 0);
  EL(silu_sum_kernel, BATCH * EMDD, st, temb, lemb, BATCH * EMDD);

  // -------- 2) global_emb = cheb_conv(cemd) --------
  EL(lx_kernel, BATCH * N_PTS * FEATC, cemd, cemdL, Lm, BATCH, FEATC);
  gemm(stream, cemd,  P(P_GIN1_W),                nullptr,     globalE, ROWS, HIDD, FEATC, FEATC, HIDD, HIDD, 0, 0);
  gemm(stream, cemdL, P(P_GIN1_W) + FEATC * HIDD, P(P_GIN1_B), globalE, ROWS, HIDD, FEATC, FEATC, HIDD, HIDD, 1, 0);

  // -------- 3) local branch: top-k radar points + local attention --------
  topk_kernel<<<ROWS, 128, 0, stream>>>(x, radar, selIdx, scaleArr);
  EL(gather_pcs_kernel, LROWS * 6, radar, selIdx, pcs);
  gemm(stream, pcs, P(P_LOC_W0), P(P_LOC_B0), leh, LROWS, 48, 6, 6, 48, 48, 0, 2);
  gemm(stream, leh, P(P_LOC_W1), P(P_LOC_B1), le,  LROWS, HIDD, 48, 48, HIDD, HIDD, 0, 0);
  for (int i = 0; i < 5; ++i) {
    const float* w = P(P_LA_W) + (size_t)i * 4 * HIDD * HIDD;
    const float* b = P(P_LA_B) + (size_t)i * 4 * HIDD;
    gemm(stream, le, w + 0 * HIDD * HIDD, b + 0 * HIDD, qb,   LROWS, HIDD, HIDD, HIDD, HIDD, HIDD, 0, 0);
    gemm(stream, le, w + 1 * HIDD * HIDD, b + 1 * HIDD, kbuf, LROWS, HIDD, HIDD, HIDD, HIDD, HIDD, 0, 0);
    gemm(stream, le, w + 2 * HIDD * HIDD, b + 2 * HIDD, vb,   LROWS, HIDD, HIDD, HIDD, HIDD, HIDD, 0, 0);
    attn_kernel<<<dim3(ROWS, HEADS), 64, 0, stream>>>(qb, kbuf, vb, ao, K_NEAR);
    gemm(stream, ao, w + 3 * HIDD * HIDD, b + 3 * HIDD, le, LROWS, HIDD, HIDD, HIDD, HIDD, HIDD, 1, 0);
  }
  EL(meanscale_kernel, ROWS * HIDD, le, scaleArr, localE);

  // -------- 4) past branch: cheb on frames + 2 temporal convs (im2col GEMM) --------
  EL(frames_kernel, BATCH * PASTF * N_PTS * 3, x, frames);
  EL(lx_kernel, RPROWS * N_PTS * 3, frames, framesL, Lm, RPROWS, 3);
  gemm(stream, frames,  P(P_GINT_W),            nullptr,     xp, RPROWS * N_PTS, HIDD, 3, 3, HIDD, HIDD, 0, 0);
  gemm(stream, framesL, P(P_GINT_W) + 3 * HIDD, P(P_GINT_B), xp, RPROWS * N_PTS, HIDD, 3, 3, HIDD, HIDD, 1, 0);
  // xp is exactly (RPROWS x THID) row-major
  EL(repack_w_kernel, THIDC * THIDC * 3, P(P_TC1_W), wp);
  EL(im2col_kernel, RPROWS * KIM, xp, colb, RPROWS, THIDC);
  gemm(stream, colb, wp, P(P_TC1_B), pe1, RPROWS, THIDC, KIM, KIM, THIDC, THIDC, 0, 1);
  EL(repack_w_kernel, THIDC * THIDC * 3, P(P_TC2_W), wp);
  EL(im2col_kernel, RPROWS * KIM, pe1, colb, RPROWS, THIDC);
  gemm(stream, colb, wp, P(P_TC2_B), pe2, RPROWS, THIDC, KIM, KIM, THIDC, THIDC, 0, 0);
  EL(maxtime_kernel, ROWS * HIDD, pe2, pastE);

  // -------- 5) cond + limb prediction --------
  EL(add3_kernel, ROWS * HIDD, cond, localE, globalE, pastE, ROWS * HIDD);
  ln_torch_kernel<<<BATCH, 32, 0, stream>>>(cemd, P(P_LL_G), P(P_LL_B), lnC, LLD);
  gemm(stream, lnC, P(P_LL_W0), P(P_LL_B0), llh, BATCH, LIMBF, LLD, LLD, LIMBF, LIMBF, 0, 1);
  gemm(stream, llh, P(P_LL_W1), P(P_LL_B1), out_limb, BATCH, LIMB, LIMBF, LIMBF, LIMB, LIMB, 0, 0);

  // -------- 6) out = cheb_conv(x[...,-3:]) + cond --------
  EL(xlast_kernel, ROWS * 3, x, xlast);
  EL(lx_kernel, BATCH * N_PTS * 3, xlast, xlastL, Lm, BATCH, 3);
  EL(copy_kernel, ROWS * HIDD, outb, cond, ROWS * HIDD);
  gemm(stream, xlast,  P(P_GIN2_W),            nullptr,     outb, ROWS, HIDD, 3, 3, HIDD, HIDD, 1, 0);
  gemm(stream, xlastL, P(P_GIN2_W) + 3 * HIDD, P(P_GIN2_B), outb, ROWS, HIDD, 3, 3, HIDD, HIDD, 1, 0);

  // -------- 7) main trunk layers --------
  for (int i = 0; i < LAYERS; ++i) {
    const float* aw = P(P_AW) + (size_t)i * 4 * HIDD * HIDD;
    const float* ab = P(P_AB) + (size_t)i * 4 * HIDD;
    // self-attention with pre-LN (mask is all ones)
    ln_gra_kernel<<<ROWS, 32, 0, stream>>>(outb, P(P_LN1_G) + i * HIDD, P(P_LN1_B) + i * HIDD, hbuf, HIDD);
    gemm(stream, hbuf, aw + 0 * HIDD * HIDD, ab + 0 * HIDD, qb,   ROWS, HIDD, HIDD, HIDD, HIDD, HIDD, 0, 0);
    gemm(stream, hbuf, aw + 1 * HIDD * HIDD, ab + 1 * HIDD, kbuf, ROWS, HIDD, HIDD, HIDD, HIDD, HIDD, 0, 0);
    gemm(stream, hbuf, aw + 2 * HIDD * HIDD, ab + 2 * HIDD, vb,   ROWS, HIDD, HIDD, HIDD, HIDD, HIDD, 0, 0);
    attn_kernel<<<dim3(BATCH, HEADS), 64, 0, stream>>>(qb, kbuf, vb, ao, N_PTS);
    gemm(stream, ao, aw + 3 * HIDD * HIDD, ab + 3 * HIDD, outb, ROWS, HIDD, HIDD, HIDD, HIDD, HIDD, 1, 0);
    // linear with pre-LN
    ln_gra_kernel<<<ROWS, 32, 0, stream>>>(outb, P(P_LN2_G) + i * HIDD, P(P_LN2_B) + i * HIDD, hbuf, HIDD);
    gemm(stream, hbuf, P(P_GCN_W) + (size_t)i * HIDD * HIDD, P(P_GCN_B) + i * HIDD, outb,
         ROWS, HIDD, HIDD, HIDD, HIDD, HIDD, 1, 0);
    // h = relu(cheb(out, g1)) ; h += tp(silu(temb_all)) + cond ; h = relu(cheb(h, g2)) ; out += h
    const float* g1 = P(P_G1_W) + (size_t)i * 2 * HIDD * HIDD;
    const float* g2 = P(P_G2_W) + (size_t)i * 2 * HIDD * HIDD;
    EL(lx_kernel, BATCH * N_PTS * HIDD, outb, Lbuf, Lm, BATCH, HIDD);
    gemm(stream, outb, g1,               nullptr,              hbuf, ROWS, HIDD, HIDD, HIDD, HIDD, HIDD, 0, 0);
    gemm(stream, Lbuf, g1 + HIDD * HIDD, P(P_G1_B) + i * HIDD, hbuf, ROWS, HIDD, HIDD, HIDD, HIDD, HIDD, 1, 1);
    gemm(stream, st, P(P_TP_W) + (size_t)i * EMDD * HIDD, P(P_TP_B) + i * HIDD, tvec,
         BATCH, HIDD, EMDD, EMDD, HIDD, HIDD, 0, 0);
    EL(addbc_kernel, ROWS * HIDD, hbuf, tvec, cond, ROWS * HIDD);
    EL(lx_kernel, BATCH * N_PTS * HIDD, hbuf, Lbuf, Lm, BATCH, HIDD);
    gemm(stream, hbuf, g2,               nullptr,              h2, ROWS, HIDD, HIDD, HIDD, HIDD, HIDD, 0, 0);
    gemm(stream, Lbuf, g2 + HIDD * HIDD, P(P_G2_B) + i * HIDD, h2, ROWS, HIDD, HIDD, HIDD, HIDD, HIDD, 1, 1);
    EL(addin_kernel, ROWS * HIDD, outb, h2, ROWS * HIDD);
  }

  // -------- 8) final cheb_conv -> pose output --------
  EL(lx_kernel, BATCH * N_PTS * HIDD, outb, Lbuf, Lm, BATCH, HIDD);
  gemm(stream, outb, P(P_GO_W),               nullptr,    out_pose, ROWS, XDIM, HIDD, HIDD, XDIM, XDIM, 0, 0);
  gemm(stream, Lbuf, P(P_GO_W) + HIDD * XDIM, P(P_GO_B),  out_pose, ROWS, XDIM, HIDD, HIDD, XDIM, XDIM, 1, 0);
}
